// SingleLayerMoEScheme3_42700564857410
// MI455X (gfx1250) — compile-verified, matrix operand order checked by reading, not run on hardware
//
#include <hip/hip_runtime.h>
#include <hip/hip_bf16.h>

#define N_TOK 4096
#define DIM   512
#define HID   2048
#define NEXP  8
#define NC    7
#define MT    32                 // tokens per MoE block (two 16-row WMMA tiles)
#define HSPLIT 2                 // H-reduction split across blocks (partial sums via f32 atomics)
#define BETA_FIXED (1.0f/3.0f)
#define BETA_DYN   (2.0f/3.0f)

typedef __attribute__((ext_vector_type(16))) __bf16 v16bf_t;
typedef __attribute__((ext_vector_type(8)))  __bf16 v8bf_t;
typedef __attribute__((ext_vector_type(8)))  float  v8f_t;

// Build a 16x32 bf16 A/B WMMA fragment slice for this lane:
// 8 contiguous bf16 at p, plus 8 contiguous bf16 at p+16 (ISA 16-bit layout).
__device__ __forceinline__ v16bf_t frag16(const __bf16* p) {
  v8bf_t lo = *reinterpret_cast<const v8bf_t*>(p);
  v8bf_t hi = *reinterpret_cast<const v8bf_t*>(p + 16);
  return __builtin_shufflevector(lo, hi, 0,1,2,3,4,5,6,7,8,9,10,11,12,13,14,15);
}

// ---------------- init: zero output accumulation region + scratch ----------------
__global__ __launch_bounds__(256) void init_zero(float* out, float* combine,
                                                 int* counts, float* imp) {
  int i = blockIdx.x * 256 + threadIdx.x;
  if (i < N_TOK * DIM) out[i] = 0.0f;          // mixed accumulator
  if (i < N_TOK * NEXP) combine[i] = 0.0f;     // per-(token,expert) gate weights
  if (i < NEXP) counts[i] = 0;
  if (i < NC)   imp[i] = 0.0f;
}

// ---------------- fp32 -> bf16 transpose (per-expert [R,C] -> [C,R]) ----------------
__global__ __launch_bounds__(256) void transpose_bf16(const float* __restrict__ src,
                                                      __bf16* __restrict__ dst,
                                                      int R, int C) {
  __shared__ __bf16 tile[32][33];
  const int e = blockIdx.z;
  const float* s = src + (size_t)e * R * C;
  __bf16*      d = dst + (size_t)e * R * C;
  const int c0 = blockIdx.x * 32, r0 = blockIdx.y * 32;
  const int tx = threadIdx.x, ty = threadIdx.y;
  #pragma unroll
  for (int j = 0; j < 4; ++j) {
    int r = r0 + ty + j * 8;
    tile[ty + j * 8][tx] = (__bf16)s[(size_t)r * C + c0 + tx];
  }
  __syncthreads();
  #pragma unroll
  for (int j = 0; j < 4; ++j) {
    int c = c0 + ty + j * 8;
    d[(size_t)c * R + r0 + tx] = tile[tx][ty + j * 8];
  }
}

// ---------------- x fp32 -> bf16 ----------------
__global__ __launch_bounds__(256) void convert_x(const float* __restrict__ x,
                                                 __bf16* __restrict__ xb) {
  int i = blockIdx.x * 256 + threadIdx.x;
  if (i < N_TOK * DIM) xb[i] = (__bf16)x[i];
}

// ---------------- router: one wave32 per token ----------------
__global__ __launch_bounds__(256) void router_kernel(const float* __restrict__ x,
                                                     const float* __restrict__ rW,
                                                     const float* __restrict__ rb,
                                                     float* __restrict__ out,
                                                     float* __restrict__ combine,
                                                     int* __restrict__ lists,
                                                     int* __restrict__ counts,
                                                     float* __restrict__ imp) {
  const int lane = threadIdx.x & 31;
  const int wave = threadIdx.x >> 5;
  const int n = blockIdx.x * 8 + wave;
  if (n >= N_TOK) return;

  float part[NC];
  #pragma unroll
  for (int j = 0; j < NC; ++j) part[j] = 0.0f;
  for (int k = lane; k < DIM; k += 32) {
    float xv = x[(size_t)n * DIM + k];
    const float* wr = rW + (size_t)k * NC;
    #pragma unroll
    for (int j = 0; j < NC; ++j) part[j] += xv * wr[j];
  }
  #pragma unroll
  for (int j = 0; j < NC; ++j)
    for (int off = 16; off >= 1; off >>= 1)
      part[j] += __shfl_xor(part[j], off, 32);

  if (lane == 0) {
    float lg[NC];
    #pragma unroll
    for (int j = 0; j < NC; ++j) lg[j] = part[j] + rb[j];
    // top-2 (descending, tie -> lower index first, matching lax.top_k)
    int i1 = 0;
    #pragma unroll
    for (int j = 1; j < NC; ++j) if (lg[j] > lg[i1]) i1 = j;
    int i2 = (i1 == 0) ? 1 : 0;
    #pragma unroll
    for (int j = 0; j < NC; ++j) if (j != i1 && lg[j] > lg[i2]) i2 = j;
    float e2 = __expf(lg[i2] - lg[i1]);
    float inv = 1.0f / (1.0f + e2);
    combine[n * NEXP + (i1 + 1)] = BETA_DYN * inv;
    combine[n * NEXP + (i2 + 1)] = BETA_DYN * e2 * inv;
    int p1 = atomicAdd(&counts[i1 + 1], 1); lists[i1 * N_TOK + p1] = n;
    int p2 = atomicAdd(&counts[i2 + 1], 1); lists[i2 * N_TOK + p2] = n;
    // sel_expert output (stored as float values in the f32 output buffer)
    out[(size_t)N_TOK * DIM + 2 + n * 2 + 0] = (float)(i1 + 1);
    out[(size_t)N_TOK * DIM + 2 + n * 2 + 1] = (float)(i2 + 1);
    // full softmax for importance accumulation
    float mx = lg[0];
    #pragma unroll
    for (int j = 1; j < NC; ++j) mx = fmaxf(mx, lg[j]);
    float ex[NC], s = 0.0f;
    #pragma unroll
    for (int j = 0; j < NC; ++j) { ex[j] = __expf(lg[j] - mx); s += ex[j]; }
    float rs = 1.0f / s;
    #pragma unroll
    for (int j = 0; j < NC; ++j) atomicAdd(&imp[j], ex[j] * rs);
  }
}

// ---------------- aux-loss scalars ----------------
__global__ void finalize_kernel(const float* __restrict__ imp_acc, float* out) {
  if (threadIdx.x == 0 && blockIdx.x == 0) {
    float lb = 0.0f, ent = 0.0f;
    const float uni = 1.0f / NC;
    for (int j = 0; j < NC; ++j) {
      float p = imp_acc[j] / (float)N_TOK;
      float d = p - uni;
      lb += d * d;
      float pc = p < 1e-8f ? 1e-8f : p;
      ent -= p * logf(pc);
    }
    out[(size_t)N_TOK * DIM + 0] = lb / NC;
    out[(size_t)N_TOK * DIM + 1] = ent;
  }
}

// ---------------- fused MoE: (expert, 32-token tile, H-split) per block, 8 wave32s ----
// Two 16-row WMMA tiles share every B fragment -> 2 WMMAs per weight load.
// The H reduction is split across HSPLIT blocks: each covers HID/HSPLIT hidden
// units (half of W1's columns, half of W2's rows) and scatter-adds its partial
// w*y into the zero-initialized f32 output. Doubles concurrency at constant
// total L2 weight traffic.
__global__ __launch_bounds__(256) void moe_kernel(const __bf16* __restrict__ xb,
                                                  const __bf16* __restrict__ w1t, // [E,H,D]
                                                  const __bf16* __restrict__ w2t, // [E,D,H]
                                                  const float* __restrict__ b1,   // [E,H]
                                                  const float* __restrict__ b2,   // [E,D]
                                                  const float* __restrict__ combine,
                                                  const int* __restrict__ lists,
                                                  const int* __restrict__ counts,
                                                  const float* __restrict__ fixed_w,
                                                  float* __restrict__ out) {
  const int ntiles = N_TOK / MT;              // 128
  const int hsp  = blockIdx.x % HSPLIT;
  const int tile = (blockIdx.x / HSPLIT) % ntiles;
  const int e    = blockIdx.x / (HSPLIT * ntiles);
  const int cnt  = (e == 0) ? N_TOK : counts[e];
  if (tile * MT >= cnt) return;               // compaction: unused (expert,tile) exits

  __shared__ __bf16 xs[MT][DIM];              // 32 KB gathered token tile
  __shared__ __bf16 hs[MT][128];              // 8 KB hidden chunk
  __shared__ int    toks[MT];
  __shared__ float  wrow[MT];

  const int lane = threadIdx.x & 31;
  const int wave = threadIdx.x >> 5;
  const int mrow = lane & 15;
  const int koff = (lane & 16) ? 8 : 0;
  const int rbase = (lane & 16) ? 8 : 0;

  if (threadIdx.x < MT) {
    int idx = tile * MT + threadIdx.x;
    int n = 0; float w = 0.0f;
    if (idx < cnt) {
      n = (e == 0) ? idx : lists[(e - 1) * N_TOK + idx];
      w = (e == 0) ? (BETA_FIXED * fixed_w[0]) : combine[n * NEXP + e];
    }
    toks[threadIdx.x] = n;
    wrow[threadIdx.x] = w;
  }
  __syncthreads();

  // stage gathered x tile: MT*DIM/8 16-byte chunks across 256 threads
  for (int i = threadIdx.x; i < MT * DIM / 8; i += 256) {
    int row = i >> 6;             // 64 chunks per row
    int col = (i & 63) << 3;
    *reinterpret_cast<v8bf_t*>(&xs[row][col]) =
        *reinterpret_cast<const v8bf_t*>(xb + (size_t)toks[row] * DIM + col);
  }
  __syncthreads();

  v8f_t acc[2][4];                            // [row tile][D tile]
  #pragma unroll
  for (int t = 0; t < 2; ++t)
    #pragma unroll
    for (int j = 0; j < 4; ++j)
      #pragma unroll
      for (int i = 0; i < 8; ++i) acc[t][j][i] = 0.0f;

  const int nchunks = HID / 128 / HSPLIT;     // 8 chunks per split
  const int hc0 = hsp * nchunks;
  for (int hc = hc0; hc < hc0 + nchunks; ++hc) {
    // ---- layer 1: this wave computes 32 tokens x 16 hidden cols, K = 512 ----
    v8f_t h1a, h1b;
    #pragma unroll
    for (int i = 0; i < 8; ++i) { h1a[i] = 0.0f; h1b[i] = 0.0f; }
    const int ncol = hc * 128 + wave * 16 + mrow;
    const __bf16* bptr = w1t + ((size_t)e * HID + ncol) * DIM + koff;

    v16bf_t bnext = frag16(bptr);             // one-step register prefetch
    #pragma unroll
    for (int k0 = 0; k0 < DIM; k0 += 32) {
      v16bf_t bcur = bnext;
      if (k0 + 32 < DIM) bnext = frag16(bptr + k0 + 32);
      v16bf_t a0 = frag16(&xs[mrow][k0 + koff]);
      v16bf_t a1 = frag16(&xs[16 + mrow][k0 + koff]);
      h1a = __builtin_amdgcn_wmma_f32_16x16x32_bf16(false, a0, false, bcur,
                                                    (short)0, h1a, false, false);
      h1b = __builtin_amdgcn_wmma_f32_16x16x32_bf16(false, a1, false, bcur,
                                                    (short)0, h1b, false, false);
    }
    const float bias1 = b1[(size_t)e * HID + ncol];
    #pragma unroll
    for (int v = 0; v < 8; ++v) {
      float va = h1a[v] + bias1;
      float vb = h1b[v] + bias1;
      va = va > 0.0f ? va : 0.0f;              // ReLU
      vb = vb > 0.0f ? vb : 0.0f;
      hs[rbase + v][wave * 16 + mrow] = (__bf16)va;
      hs[16 + rbase + v][wave * 16 + mrow] = (__bf16)vb;
    }
    __syncthreads();

    // ---- layer 2 partial: this wave owns 32 tokens x 64 out cols, K-chunk = 128 ----
    v16bf_t afr[2][4];
    #pragma unroll
    for (int t = 0; t < 2; ++t)
      #pragma unroll
      for (int kk = 0; kk < 4; ++kk)
        afr[t][kk] = frag16(&hs[t * 16 + mrow][kk * 32 + koff]);
    #pragma unroll
    for (int j = 0; j < 4; ++j) {
      const int dcol = wave * 64 + j * 16 + mrow;
      const __bf16* b2p = w2t + ((size_t)e * DIM + dcol) * HID + hc * 128 + koff;
      #pragma unroll
      for (int kk = 0; kk < 4; ++kk) {
        v16bf_t b = frag16(b2p + kk * 32);
        acc[0][j] = __builtin_amdgcn_wmma_f32_16x16x32_bf16(false, afr[0][kk], false, b,
                                                            (short)0, acc[0][j], false, false);
        acc[1][j] = __builtin_amdgcn_wmma_f32_16x16x32_bf16(false, afr[1][kk], false, b,
                                                            (short)0, acc[1][j], false, false);
      }
    }
    __syncthreads();
  }

  // ---- epilogue: +b2 (split 0 only), scale by combine weight, scatter-add ----
  #pragma unroll
  for (int j = 0; j < 4; ++j) {
    const int dcol = wave * 64 + j * 16 + mrow;
    const float bias2 = (hsp == 0) ? b2[(size_t)e * DIM + dcol] : 0.0f;
    #pragma unroll
    for (int t = 0; t < 2; ++t) {
      #pragma unroll
      for (int v = 0; v < 8; ++v) {
        int row = t * 16 + rbase + v;
        float val = (acc[t][j][v] + bias2) * wrow[row];
        atomicAdd(out + (size_t)toks[row] * DIM + dcol, val);
      }
    }
  }
}

extern "C" void kernel_launch(void* const* d_in, const int* in_sizes, int n_in,
                              void* d_out, int out_size, void* d_ws, size_t ws_size,
                              hipStream_t stream) {
  (void)in_sizes; (void)n_in; (void)out_size; (void)ws_size;
  const float* x   = (const float*)d_in[0];
  const float* rW  = (const float*)d_in[1];
  const float* rb  = (const float*)d_in[2];
  const float* W1  = (const float*)d_in[3];
  const float* b1  = (const float*)d_in[4];
  const float* W2  = (const float*)d_in[5];
  const float* b2  = (const float*)d_in[6];
  const float* fxw = (const float*)d_in[7];
  float* out = (float*)d_out;

  char* ws = (char*)d_ws;
  __bf16* W1T   = (__bf16*)(ws);                       // 16 MB  [E,H,D]
  __bf16* W2T   = (__bf16*)(ws + 16777216);            // 16 MB  [E,D,H]
  __bf16* XB    = (__bf16*)(ws + 33554432);            // 4 MB   [N,D]
  float*  COMB  = (float*)(ws + 37748736);             // 128 KB [N,E]
  int*    LISTS = (int*)  (ws + 37879808);             // 112 KB [NC,N]
  int*    COUNTS= (int*)  (ws + 37994496);             // 32 B
  float*  IMP   = (float*)(ws + 37994528);             // 28 B

  init_zero<<<(N_TOK * DIM) / 256, 256, 0, stream>>>(out, COMB, COUNTS, IMP);

  dim3 tb(32, 8);
  transpose_bf16<<<dim3(HID / 32, DIM / 32, NEXP), tb, 0, stream>>>(W1, W1T, DIM, HID);
  transpose_bf16<<<dim3(DIM / 32, HID / 32, NEXP), tb, 0, stream>>>(W2, W2T, HID, DIM);
  convert_x<<<(N_TOK * DIM) / 256, 256, 0, stream>>>(x, XB);

  router_kernel<<<N_TOK / 8, 256, 0, stream>>>(x, rW, rb, out, COMB, LISTS, COUNTS, IMP);
  finalize_kernel<<<1, 32, 0, stream>>>(IMP, out);

  moe_kernel<<<NEXP * (N_TOK / MT) * HSPLIT, 256, 0, stream>>>(XB, W1T, W2T, b1, b2,
                                                               COMB, LISTS, COUNTS, fxw, out);
}
